// PalmDecoderSelfAttention_31971736552008
// MI455X (gfx1250) — matrix operand order repaired and checked by reading, not compiled, not run on hardware
//
#include <hip/hip_runtime.h>
#include <hip/hip_bf16.h>

// ---------------------------------------------------------------------------
// Types for CDNA5 WMMA (wave32): 16x16x32 bf16 -> f32
// ---------------------------------------------------------------------------
typedef __attribute__((ext_vector_type(16))) __bf16        v16bf;
typedef __attribute__((ext_vector_type(8)))  float         v8f;
typedef __attribute__((ext_vector_type(4)))  unsigned int  u32x4;
typedef __attribute__((ext_vector_type(8)))  int           i32x8;
typedef __attribute__((ext_vector_type(4)))  int           i32x4;

typedef __attribute__((address_space(1))) void gvoid;
typedef __attribute__((address_space(3))) void lvoid;

struct B32x2 { u32x4 a, b; };

static __device__ __forceinline__ u32x4 ld16(const __bf16* p) {
    return *(const u32x4*)p;
}
static __device__ __forceinline__ v16bf make_frag(u32x4 a, u32x4 b) {
    B32x2 t{a, b};
    return __builtin_bit_cast(v16bf, t);
}
static __device__ __forceinline__ __bf16 f32_to_bf16(float f) {
    unsigned u = __builtin_bit_cast(unsigned, f);
    u += 0x7fffu + ((u >> 16) & 1u);              // round-to-nearest-even
    return __builtin_bit_cast(__bf16, (unsigned short)(u >> 16));
}
static __device__ __forceinline__ v8f wmma_bf16(v16bf a, v16bf b, v8f c) {
    // (neg_a, A, neg_b, B, c_mod, C, reuse_a, reuse_b)
    return __builtin_amdgcn_wmma_f32_16x16x32_bf16(false, a, false, b,
                                                   (short)0, c, false, false);
}
static __device__ __forceinline__ unsigned lds_addr_of(void* p) {
    return (unsigned)(unsigned long long)(lvoid*)p;   // addrspacecast -> LDS byte offset
}

// CDNA5 data-movement feature detection (never regresses compilation)
#if defined(__AMDGCN__) && __has_builtin(__builtin_amdgcn_tensor_load_to_lds) && \
    __has_builtin(__builtin_amdgcn_s_wait_tensorcnt)
#define HAS_TDM 1
#else
#define HAS_TDM 0
#endif
#if defined(__AMDGCN__) && __has_builtin(__builtin_amdgcn_global_load_async_to_lds_b128) && \
    __has_builtin(__builtin_amdgcn_s_wait_asynccnt)
#define HAS_ASYNC 1
#else
#define HAS_ASYNC 0
#endif

// ---------------------------------------------------------------------------
// Problem constants
// ---------------------------------------------------------------------------
#define BATCH 2
#define SEQ   2048
#define HID   1024
#define NH    16
#define HD    64
#define MROWS (BATCH * SEQ)   // 4096

// ---------------------------------------------------------------------------
// Kernel 1: fp32 -> bf16 elementwise
// ---------------------------------------------------------------------------
__global__ void convert_bf16_kernel(const float* __restrict__ src,
                                    __bf16* __restrict__ dst, int n) {
    int i = blockIdx.x * blockDim.x + threadIdx.x;
    if (i < n) dst[i] = f32_to_bf16(src[i]);
}

// ---------------------------------------------------------------------------
// Kernel 2: fp32 [K][N] -> bf16 transposed [N][K]
// ---------------------------------------------------------------------------
__global__ void convert_transpose_kernel(const float* __restrict__ src,
                                         __bf16* __restrict__ dst,
                                         int K, int N) {
    int i = blockIdx.x * blockDim.x + threadIdx.x;
    if (i < K * N) {
        int k = i / N, n = i - k * N;
        dst[n * K + k] = f32_to_bf16(src[i]);
    }
}

// ---------------------------------------------------------------------------
// Kernel 3: QKV GEMM.  C[M=4096, N=3072] = Xbf @ Wqkv + b, scattered into
// head-major Q/K/V [B, NH, S, HD] bf16 buffers.
// Block = 128 threads (4 waves); block tile 128x64, wave tile 32x64
// (acc[2][4] => 8 WMMA per k-step vs 12 fragment loads).
// ---------------------------------------------------------------------------
__global__ __launch_bounds__(128) void gemm_qkv_kernel(
    const __bf16* __restrict__ X,   // [4096][1024] bf16 row-major
    const __bf16* __restrict__ Wt,  // [3072][1024] bf16 (N-major)
    const float*  __restrict__ bias,// [3072]
    __bf16* __restrict__ Qh, __bf16* __restrict__ Kh, __bf16* __restrict__ Vh)
{
    const int lane = threadIdx.x & 31;
    const int w    = threadIdx.x >> 5;
    const int lr   = lane & 15;
    const int half = lane >> 4;
    const int Mb   = blockIdx.y * 128 + w * 32;
    const int Nb   = blockIdx.x * 64;

    v8f acc[2][4];
#pragma unroll
    for (int mt = 0; mt < 2; ++mt)
#pragma unroll
        for (int nt = 0; nt < 4; ++nt) acc[mt][nt] = {};

    for (int kk = 0; kk < HID; kk += 32) {
        v16bf a[2], b[4];
#pragma unroll
        for (int mt = 0; mt < 2; ++mt) {
            const __bf16* p = X + (size_t)(Mb + mt * 16 + lr) * HID + kk + half * 8;
            a[mt] = make_frag(ld16(p), ld16(p + 16));
        }
#pragma unroll
        for (int nt = 0; nt < 4; ++nt) {
            const __bf16* p = Wt + (size_t)(Nb + nt * 16 + lr) * HID + kk + half * 8;
            b[nt] = make_frag(ld16(p), ld16(p + 16));
        }
        if (kk + 32 < HID) {   // stream next tiles into cache
            __builtin_prefetch(X  + (size_t)(Mb + lr) * HID + kk + 32, 0, 1);
            __builtin_prefetch(Wt + (size_t)(Nb + lr) * HID + kk + 32, 0, 1);
        }
#pragma unroll
        for (int mt = 0; mt < 2; ++mt)
#pragma unroll
            for (int nt = 0; nt < 4; ++nt)
                acc[mt][nt] = wmma_bf16(a[mt], b[nt], acc[mt][nt]);
    }

    // Epilogue: bias add + scatter to head-major Q/K/V (bf16)
#pragma unroll
    for (int nt = 0; nt < 4; ++nt) {
        const int colg = Nb + nt * 16 + lr;           // [0, 3072)
        const float bv = bias[colg];
        const int which = colg >> 10;                 // 0:q 1:k 2:v
        const int hcol  = colg & 1023;
        const int head  = hcol >> 6;
        const int hdi   = hcol & 63;
        __bf16* dst = (which == 0) ? Qh : (which == 1) ? Kh : Vh;
#pragma unroll
        for (int mt = 0; mt < 2; ++mt)
#pragma unroll
            for (int r = 0; r < 8; ++r) {
                const int row = Mb + mt * 16 + r + half * 8;
                const int bb = row >> 11, ss = row & 2047;
                dst[(((size_t)(bb * NH + head)) * SEQ + ss) * HD + hdi] =
                    f32_to_bf16(acc[mt][nt][r] + bv);
            }
    }
}

// ---------------------------------------------------------------------------
// Kernel 4: flash attention (causal), bf16 WMMA, f32 online softmax.
// Grid: (S/64, NH, B).  Block = 128 thr (4 waves); wave handles 16 q rows.
// K tile staged via TDM (tensor_load_to_lds) or async-to-LDS when available.
// LDS: K tile [32][64], V tile transposed [64][32], per-wave P tile [16][32].
// ---------------------------------------------------------------------------
__global__ __launch_bounds__(128) void attn_kernel(
    const __bf16* __restrict__ Qh, const __bf16* __restrict__ Kh,
    const __bf16* __restrict__ Vh, __bf16* __restrict__ Ctx)
{
    __shared__ __bf16 sK[32 * 64];
    __shared__ __bf16 sVt[64 * 32];
    __shared__ __bf16 sP[4][16 * 32];

    const int lane = threadIdx.x & 31;
    const int w    = threadIdx.x >> 5;
    const int lr   = lane & 15;
    const int half = lane >> 4;
    const int b    = blockIdx.z;
    const int h    = blockIdx.y;
    const int qt   = blockIdx.x;

    const size_t headoff = ((size_t)(b * NH + h)) * SEQ * HD;
    const __bf16* Q = Qh + headoff;
    const __bf16* K = Kh + headoff;
    const __bf16* V = Vh + headoff;

    const int qbase = qt * 64 + w * 16;

    // Q A-fragments for hd = 0..31 and 32..63 (loaded once, reused all steps)
    v16bf qa[2];
    {
        const __bf16* p = Q + (size_t)(qbase + lr) * HD + half * 8;
        qa[0] = make_frag(ld16(p),      ld16(p + 16));
        qa[1] = make_frag(ld16(p + 32), ld16(p + 48));
    }

    float m[8], l[8];
    v8f ctx[4];
#pragma unroll
    for (int r = 0; r < 8; ++r) { m[r] = -1e30f; l[r] = 0.0f; }
#pragma unroll
    for (int n = 0; n < 4; ++n) ctx[n] = {};

    const int nsteps = qt * 2 + 2;     // causal: keys up to qt*64+63
    for (int step = 0; step < nsteps; ++step) {
        const int kc = step * 32;

        // ---- stage K tile [32 x 64] bf16 into LDS ----
#if HAS_TDM
        if (w == 0) {
            // Tensor DMA: 2-D tile (64 x 32 rows) of a [2048][64] bf16 tensor.
            const unsigned long long ga =
                (unsigned long long)(const void*)(K + (size_t)kc * HD);
            const unsigned lds = lds_addr_of((void*)sK);
            // D# group0: count=1, lds_addr, global_addr[56:0], type=2
            u32x4 g0 = { 1u, lds, (unsigned)ga,
                         ((unsigned)(ga >> 32) & 0x01ffffffu) | (2u << 30) };
            // D# group1: data_size=2B; tensor_dim0=64, tensor_dim1=2048;
            //            tile_dim0=64, tile_dim1=32; dim0_stride=64
            i32x8 g1 = { (int)(1u << 16),       // wg_mask=0 | data_size=1(2B)
                         (int)(64u << 16),      // tensor_dim0 lo16 << 16
                         (int)(2048u << 16),    // dim0 hi=0 | tensor_dim1 lo16
                         (int)(64u << 16),      // dim1 hi=0 | tile_dim0=64
                         32,                    // tile_dim1=32 | tile_dim2=0
                         64,                    // tensor_dim0_stride lo32
                         0, 0 };
            i32x4 z4 = {};
#if __clang_major__ >= 23
            i32x8 z8 = {};
            __builtin_amdgcn_tensor_load_to_lds(g0, g1, z4, z4, z8, 0);
#else
            __builtin_amdgcn_tensor_load_to_lds(g0, g1, z4, z4, 0);
#endif
        }
#elif HAS_ASYNC
#pragma unroll
        for (int i = 0; i < 2; ++i) {
            const int c  = threadIdx.x + i * 128;    // 256 chunks of 8 bf16
            const int rk = c >> 3;
            const int c0 = (c & 7) * 8;
            __builtin_amdgcn_global_load_async_to_lds_b128(
                (gvoid*)(K + (size_t)(kc + rk) * HD + c0),
                (lvoid*)(sK + rk * 64 + c0), 0, 0);
        }
#else
#pragma unroll
        for (int i = 0; i < 2; ++i) {
            const int c  = threadIdx.x + i * 128;
            const int rk = c >> 3;
            const int c0 = (c & 7) * 8;
            *(u32x4*)(sK + rk * 64 + c0) = ld16(K + (size_t)(kc + rk) * HD + c0);
        }
#endif

        // ---- stage V tile transposed [64 x 32] (in-register transpose) ----
#pragma unroll
        for (int i = 0; i < 2; ++i) {
            const int c  = threadIdx.x + i * 128;
            const int rk = c >> 3;
            const int c0 = (c & 7) * 8;
            u32x4 vv = ld16(V + (size_t)(kc + rk) * HD + c0);
#pragma unroll
            for (int j = 0; j < 8; ++j) {
                unsigned word = vv[j >> 1];
                unsigned short hw = (j & 1) ? (unsigned short)(word >> 16)
                                            : (unsigned short)(word & 0xffffu);
                sVt[(c0 + j) * 32 + rk] = __builtin_bit_cast(__bf16, hw);
            }
        }
#if HAS_TDM
        __builtin_amdgcn_s_wait_tensorcnt(0);   // no-op for waves with cnt==0
#elif HAS_ASYNC
        __builtin_amdgcn_s_wait_asynccnt(0);
#endif
        __syncthreads();

        // ---- scores: S[16 x 32] = Q @ K^T  (two 16x16 n-tiles) ----
        v8f s[2];
#pragma unroll
        for (int nt = 0; nt < 2; ++nt) {
            const __bf16* p = sK + (nt * 16 + lr) * 64 + half * 8;
            v16bf b0 = make_frag(ld16(p),      ld16(p + 16));
            v16bf b1 = make_frag(ld16(p + 32), ld16(p + 48));
            v8f a = {};
            a = wmma_bf16(qa[0], b0, a);
            a = wmma_bf16(qa[1], b1, a);
            s[nt] = a;
        }

        // ---- causal mask + online softmax (rows live across 16-lane halves) ----
#pragma unroll
        for (int r = 0; r < 8; ++r) {
            const int row = qbase + r + half * 8;
            const int c0 = kc + lr, c1 = kc + 16 + lr;
            float s0 = (c0 <= row) ? s[0][r] * 0.125f : -10000.0f;
            float s1 = (c1 <= row) ? s[1][r] * 0.125f : -10000.0f;
            float t = fmaxf(s0, s1);
            t = fmaxf(t, __shfl_xor(t, 1));
            t = fmaxf(t, __shfl_xor(t, 2));
            t = fmaxf(t, __shfl_xor(t, 4));
            t = fmaxf(t, __shfl_xor(t, 8));
            const float mnew = fmaxf(m[r], t);
            const float p0 = __expf(s0 - mnew);
            const float p1 = __expf(s1 - mnew);
            float rs = p0 + p1;
            rs += __shfl_xor(rs, 1);
            rs += __shfl_xor(rs, 2);
            rs += __shfl_xor(rs, 4);
            rs += __shfl_xor(rs, 8);
            const float alpha = __expf(m[r] - mnew);
            l[r] = l[r] * alpha + rs;
            m[r] = mnew;
#pragma unroll
            for (int n = 0; n < 4; ++n) ctx[n][r] *= alpha;
            sP[w][(r + half * 8) * 32 + lr]      = f32_to_bf16(p0);
            sP[w][(r + half * 8) * 32 + 16 + lr] = f32_to_bf16(p1);
        }
        __syncthreads();

        // ---- ctx += P[16x32] @ V[32x64] (4 n-tiles of 16) ----
        {
            const __bf16* p = sP[w] + lr * 32 + half * 8;
            v16bf pf = make_frag(ld16(p), ld16(p + 16));
#pragma unroll
            for (int n = 0; n < 4; ++n) {
                const __bf16* vp = sVt + (n * 16 + lr) * 32 + half * 8;
                v16bf vb = make_frag(ld16(vp), ld16(vp + 16));
                ctx[n] = wmma_bf16(pf, vb, ctx[n]);
            }
        }
        __syncthreads();
    }

    // ---- normalize and write ctx to [B, S, H] bf16 (undoes head transpose) ----
#pragma unroll
    for (int r = 0; r < 8; ++r) {
        const float inv = 1.0f / l[r];
        const int row = qbase + r + half * 8;
        const size_t base = ((size_t)(b * SEQ) + row) * HID + h * HD;
#pragma unroll
        for (int n = 0; n < 4; ++n)
            Ctx[base + n * 16 + lr] = f32_to_bf16(ctx[n][r] * inv);
    }
}

// ---------------------------------------------------------------------------
// Kernel 5: output GEMM.  out[4096,1024] = Ctx(bf16) @ Wdense + b  (f32 out)
// Block tile 128x64, wave tile 32x64.
// ---------------------------------------------------------------------------
__global__ __launch_bounds__(128) void gemm_out_kernel(
    const __bf16* __restrict__ A,   // [4096][1024] bf16
    const __bf16* __restrict__ Wt,  // [1024][1024] bf16 (N-major)
    const float*  __restrict__ bias,// [1024]
    float* __restrict__ out)        // [4096][1024] f32
{
    const int lane = threadIdx.x & 31;
    const int w    = threadIdx.x >> 5;
    const int lr   = lane & 15;
    const int half = lane >> 4;
    const int Mb   = blockIdx.y * 128 + w * 32;
    const int Nb   = blockIdx.x * 64;

    v8f acc[2][4];
#pragma unroll
    for (int mt = 0; mt < 2; ++mt)
#pragma unroll
        for (int nt = 0; nt < 4; ++nt) acc[mt][nt] = {};

    for (int kk = 0; kk < HID; kk += 32) {
        v16bf a[2], b[4];
#pragma unroll
        for (int mt = 0; mt < 2; ++mt) {
            const __bf16* p = A + (size_t)(Mb + mt * 16 + lr) * HID + kk + half * 8;
            a[mt] = make_frag(ld16(p), ld16(p + 16));
        }
#pragma unroll
        for (int nt = 0; nt < 4; ++nt) {
            const __bf16* p = Wt + (size_t)(Nb + nt * 16 + lr) * HID + kk + half * 8;
            b[nt] = make_frag(ld16(p), ld16(p + 16));
        }
        if (kk + 32 < HID) {
            __builtin_prefetch(A  + (size_t)(Mb + lr) * HID + kk + 32, 0, 1);
            __builtin_prefetch(Wt + (size_t)(Nb + lr) * HID + kk + 32, 0, 1);
        }
#pragma unroll
        for (int mt = 0; mt < 2; ++mt)
#pragma unroll
            for (int nt = 0; nt < 4; ++nt)
                acc[mt][nt] = wmma_bf16(a[mt], b[nt], acc[mt][nt]);
    }

#pragma unroll
    for (int nt = 0; nt < 4; ++nt) {
        const int colg = Nb + nt * 16 + lr;
        const float bv = bias[colg];
#pragma unroll
        for (int mt = 0; mt < 2; ++mt)
#pragma unroll
            for (int r = 0; r < 8; ++r) {
                const int row = Mb + mt * 16 + r + half * 8;
                out[(size_t)row * HID + colg] = acc[mt][nt][r] + bv;
            }
    }
}

// ---------------------------------------------------------------------------
// Host-side launcher
// ---------------------------------------------------------------------------
extern "C" void kernel_launch(void* const* d_in, const int* in_sizes, int n_in,
                              void* d_out, int out_size, void* d_ws, size_t ws_size,
                              hipStream_t stream) {
    const float* hs    = (const float*)d_in[0];  // [B,S,H]
    // d_in[1] = ltor_mask: exact lower-triangular -> applied analytically
    const float* Wqkv  = (const float*)d_in[2];  // [H, 3H]
    const float* bqkv  = (const float*)d_in[3];  // [3H]
    const float* Wd    = (const float*)d_in[4];  // [H, H]
    const float* bd    = (const float*)d_in[5];  // [H]
    float* out = (float*)d_out;

    char* ws = (char*)d_ws;
    __bf16* Xbf    = (__bf16*)ws;  ws += (size_t)MROWS * HID * 2;        // 8 MB
    __bf16* WqkvT  = (__bf16*)ws;  ws += (size_t)3 * HID * HID * 2;      // 6 MB
    __bf16* WdT    = (__bf16*)ws;  ws += (size_t)HID * HID * 2;          // 2 MB
    __bf16* Qhd    = (__bf16*)ws;  ws += (size_t)MROWS * HID * 2;        // 8 MB
    __bf16* Khd    = (__bf16*)ws;  ws += (size_t)MROWS * HID * 2;        // 8 MB
    __bf16* Vhd    = (__bf16*)ws;  ws += (size_t)MROWS * HID * 2;        // 8 MB
    __bf16* Ctx    = (__bf16*)ws;  ws += (size_t)MROWS * HID * 2;        // 8 MB

    const int nX = MROWS * HID;            // 4,194,304
    convert_bf16_kernel<<<(nX + 255) / 256, 256, 0, stream>>>(hs, Xbf, nX);

    const int nWq = HID * 3 * HID;         // 3,145,728
    convert_transpose_kernel<<<(nWq + 255) / 256, 256, 0, stream>>>(Wqkv, WqkvT, HID, 3 * HID);
    const int nWd = HID * HID;             // 1,048,576
    convert_transpose_kernel<<<(nWd + 255) / 256, 256, 0, stream>>>(Wd, WdT, HID, HID);

    gemm_qkv_kernel<<<dim3(3 * HID / 64, MROWS / 128), 128, 0, stream>>>(
        Xbf, WqkvT, bqkv, Qhd, Khd, Vhd);

    attn_kernel<<<dim3(SEQ / 64, NH, BATCH), 128, 0, stream>>>(Qhd, Khd, Vhd, Ctx);

    gemm_out_kernel<<<dim3(HID / 64, MROWS / 128), 128, 0, stream>>>(
        Ctx, WdT, bd, out);
}
